// DenseGATLayer_15599321219367
// MI455X (gfx1250) — matrix operand order repaired and checked by reading, not compiled, not run on hardware
//
#include <hip/hip_runtime.h>
#include <stdint.h>

// ---------------------------------------------------------------------------
// MI455X fused GAT layer.
//   traffic: adj (256MB) dominates -> flash-style fusion, never materialize att
//   compute: att@Wh (17.2 GFLOP) + h@W^T on v_wmma_f32_16x16x32_bf16
//   adj staging: TDM tensor_load_to_lds, double buffered, s_wait_tensorcnt
//   softmax: static per-row shift M_i = leaky(s1_i + max(s2)) -- provably safe
//            upper bound, removes all online-rescale VALU from the hot loop
// ---------------------------------------------------------------------------

typedef __attribute__((ext_vector_type(16))) __bf16 v16bf;
typedef __attribute__((ext_vector_type(8)))  float  v8f;
typedef __attribute__((ext_vector_type(4)))  unsigned int u32x4;
typedef __attribute__((ext_vector_type(8)))  int    i32x8;
typedef __attribute__((ext_vector_type(4)))  int    i32x4;

#define LOG2E 1.44269504088896340736f
#define NEGV  (-9.0e15f)
#define FMIN  (-3.0e38f)

static __device__ __forceinline__ unsigned int pack_bf16(float lo, float hi) {
  unsigned short a = __builtin_bit_cast(unsigned short, (__bf16)lo);
  unsigned short b = __builtin_bit_cast(unsigned short, (__bf16)hi);
  return (unsigned int)a | ((unsigned int)b << 16);
}

// ---------------------------------------------------------------------------
// TDM: load a 16x32 int32 tile of adj into LDS.
//  - data_size=4B, tensor 8192x8192, stride 8192
//  - pad_enable: +4 dwords after every 32 dwords -> LDS row stride 36 dwords
//    (kills ds bank conflicts, keeps 16B alignment of each row)
// ---------------------------------------------------------------------------
static __device__ __forceinline__ void tdm_load_adj(const int* adj, int i0, int jb,
                                                    int* lds_dst) {
  unsigned long long ga =
      (unsigned long long)(uintptr_t)(adj + (size_t)i0 * 8192 + (size_t)jb * 32);
  unsigned int lds = (unsigned int)(uintptr_t)lds_dst;  // low 32 bits = LDS offset
  u32x4 g0 = { 1u,                                  // count=1, user descriptor
               lds,                                 // lds_addr
               (unsigned int)(ga & 0xffffffffu),    // global_addr[31:0]
               (unsigned int)((ga >> 32) & 0x01ffffffu) | 0x80000000u }; // type=2
  i32x8 g1 = { (int)0x07120000,   // data_size=4B, pad_en, interval=32dw, amount=4dw
               (int)0x20000000,   // tensor_dim0 = 8192 (low 16 in [63:48])
               (int)0x20000000,   // tensor_dim0 hi=0 | tensor_dim1=8192 low
               (int)0x00200000,   // tensor_dim1 hi=0 | tile_dim0=32
               16,                // tile_dim1=16, tile_dim2=0
               8192,              // tensor_dim0_stride low
               0, 0 };            // stride hi, tensor_dim1_stride (2D)
  i32x4 gz = { 0, 0, 0, 0 };
#if __clang_major__ >= 23
  i32x8 gz8 = { 0, 0, 0, 0, 0, 0, 0, 0 };
  __builtin_amdgcn_tensor_load_to_lds(g0, g1, gz, gz, gz8, 0);
#else
  __builtin_amdgcn_tensor_load_to_lds(g0, g1, gz, gz, 0);
#endif
}

// ---------------------------------------------------------------------------
__global__ void zero_kernel(float* __restrict__ p, int n) {
  int i = blockIdx.x * blockDim.x + threadIdx.x;
  if (i < n) p[i] = 0.f;
}

// ---------------------------------------------------------------------------
// max over s2 (8192 values), single block
// ---------------------------------------------------------------------------
__global__ __launch_bounds__(256) void smax_kernel(const float* __restrict__ s2,
                                                   float* __restrict__ smax) {
  __shared__ float red[256];
  float m = FMIN;
  for (int i = threadIdx.x; i < 8192; i += 256) m = fmaxf(m, s2[i]);
  red[threadIdx.x] = m;
  __syncthreads();
  #pragma unroll
  for (int s = 128; s > 0; s >>= 1) {
    if ((int)threadIdx.x < s)
      red[threadIdx.x] = fmaxf(red[threadIdx.x], red[threadIdx.x + s]);
    __syncthreads();
  }
  if (threadIdx.x == 0) smax[0] = red[0];
}

// ---------------------------------------------------------------------------
// Phase A: Wh = h @ W^T via bf16 WMMA; emit WhB (B-operand swizzled bf16 copy)
// and s1/s2 reductions.  512 blocks x 8 waves; wave = feature block fb.
// ---------------------------------------------------------------------------
__global__ __launch_bounds__(256) void wh_kernel(
    const float* __restrict__ h, const float* __restrict__ W,
    const float* __restrict__ a1, const float* __restrict__ a2,
    float* __restrict__ s1, float* __restrict__ s2,
    unsigned int* __restrict__ whb) {
  const int rb   = blockIdx.x;          // 16-row block of h
  const int fb   = threadIdx.x >> 5;    // feature block (16 cols of Wh)
  const int lane = threadIdx.x & 31;
  const int mrow = lane & 15;
  const int kg   = lane >> 4;
  const int bk   = kg * 8;
  const int f0   = fb * 16;

  v8f acc = {};
  const float* hrow = h + (size_t)(rb * 16 + mrow) * 256;  // A: row = lane&15
  const float* wrow = W + (size_t)(f0 + mrow) * 256;       // B: col(feature) = lane&15

  #pragma unroll
  for (int k0 = 0; k0 < 256; k0 += 32) {
    // A layout: vec[0..7] = K bk+0..7 ; vec[8..15] = K bk+16..23
    float4 ha0 = *(const float4*)(hrow + k0 + bk);
    float4 ha1 = *(const float4*)(hrow + k0 + bk + 4);
    float4 ha2 = *(const float4*)(hrow + k0 + bk + 16);
    float4 ha3 = *(const float4*)(hrow + k0 + bk + 20);
    // B layout: vec[0..7] = K bk+0..7 ; vec[8..15] = K bk+16..23
    float4 wb0 = *(const float4*)(wrow + k0 + bk);
    float4 wb1 = *(const float4*)(wrow + k0 + bk + 4);
    float4 wb2 = *(const float4*)(wrow + k0 + bk + 16);
    float4 wb3 = *(const float4*)(wrow + k0 + bk + 20);
    float af[16] = { ha0.x,ha0.y,ha0.z,ha0.w, ha1.x,ha1.y,ha1.z,ha1.w,
                     ha2.x,ha2.y,ha2.z,ha2.w, ha3.x,ha3.y,ha3.z,ha3.w };
    float bf[16] = { wb0.x,wb0.y,wb0.z,wb0.w, wb1.x,wb1.y,wb1.z,wb1.w,
                     wb2.x,wb2.y,wb2.z,wb2.w, wb3.x,wb3.y,wb3.z,wb3.w };
    v16bf A, B;
    #pragma unroll
    for (int i = 0; i < 16; ++i) { A[i] = (__bf16)af[i]; B[i] = (__bf16)bf[i]; }
    acc = __builtin_amdgcn_wmma_f32_16x16x32_bf16(false, A, false, B,
                                                  (short)0, acc, false, false);
  }

  // --- scatter D tile into WhB, pre-swizzled as phase-B's B operand ---------
  // D lane holds rows m = kg*8+v, col n = mrow.  In B layout (table order: K = (rb&1)*16+kg*8+v)
  // this lands at laneB = kg*16 + n, dwords (rb&1)*4 + v/2.
  {
    const int jb = rb >> 1;
    const int laneB = kg * 16 + mrow;
    uint4 d = { pack_bf16(acc[0], acc[1]), pack_bf16(acc[2], acc[3]),
                pack_bf16(acc[4], acc[5]), pack_bf16(acc[6], acc[7]) };
    unsigned int* dst = whb + ((size_t)(jb * 8 + fb) * 32 + laneB) * 8 + (rb & 1) * 4;
    *(uint4*)dst = d;
  }

  // --- s1/s2: per-row dot with a1/a2, reduce 16-lane groups, atomics --------
  const float a1c = a1[f0 + mrow];
  const float a2c = a2[f0 + mrow];
  #pragma unroll
  for (int v = 0; v < 8; ++v) {
    float c1 = acc[v] * a1c;
    float c2 = acc[v] * a2c;
    #pragma unroll
    for (int off = 1; off < 16; off <<= 1) {
      c1 += __shfl_xor(c1, off, 32);
      c2 += __shfl_xor(c2, off, 32);
    }
    if (mrow == 0) {
      atomicAdd(&s1[rb * 16 + kg * 8 + v], c1);
      atomicAdd(&s2[rb * 16 + kg * 8 + v], c2);
    }
  }
}

// ---------------------------------------------------------------------------
// Phase B: fused mask + softmax (static shift) + P@Wh + ELU.
// 512 blocks (16 rows each) x 8 waves; wave w streams column blocks
// jb = w, w+8, ...  adj tiles arrive via TDM, double buffered.
// ---------------------------------------------------------------------------
#define ADJ_ROW_STRIDE 36                   // 32 data dwords + 4 TDM pad dwords
#define ADJ_BUF_DW (16 * ADJ_ROW_STRIDE)    // 576 dwords per buffer

__global__ __launch_bounds__(256) void gat_kernel(
    const int* __restrict__ adj,
    const float* __restrict__ s1g, const float* __restrict__ s2g,
    const float* __restrict__ smaxg,
    const unsigned int* __restrict__ whb,
    float* __restrict__ out) {
  __shared__ alignas(16) int adjbuf[8][2][ADJ_BUF_DW];
  __shared__ float Lrow[16];
  __shared__ float outacc[16 * 128];

  const int i0   = blockIdx.x * 16;
  const int tid  = threadIdx.x;
  const int wave = tid >> 5;
  const int lane = tid & 31;
  const int mrow = lane & 15;
  const int kg   = lane >> 4;
  const int bk   = kg * 8;

  #pragma unroll
  for (int t = 0; t < 8; ++t) outacc[tid + 256 * t] = 0.f;
  if (tid < 16) Lrow[tid] = 0.f;

  // static per-row softmax shift: M_i = leaky(s1_i + max_j s2_j) >= all e_ij
  const float s1v = s1g[i0 + mrow];        // this lane's score row
  {
  }
  const float xM  = s1v + smaxg[0];
  const float MiL = fmaxf(xM, 0.2f * xM) * LOG2E;  // M_i in log2 domain

  float l_run = 0.f;
  v8f acc[8] = {};

  const int NJ = 8192 / 32 / 8;            // 32 column tiles per wave
  tdm_load_adj(adj, i0, wave,     &adjbuf[wave][0][0]);
  tdm_load_adj(adj, i0, wave + 8, &adjbuf[wave][1][0]);

  for (int n = 0; n < NJ; ++n) {
    const int jb = wave + 8 * n;
    const int j0 = jb * 32;
    if (n + 1 < NJ) __builtin_amdgcn_s_wait_tensorcnt(1);
    else            __builtin_amdgcn_s_wait_tensorcnt(0);
    __asm__ __volatile__("" ::: "memory");

    // adj tile reads in WMMA A layout: row=mrow, K = bk+0..7 and bk+16..23
    const int* ab = &adjbuf[wave][n & 1][mrow * ADJ_ROW_STRIDE];
    int4 aA  = *(const int4*)(ab + bk);
    int4 aA2 = *(const int4*)(ab + bk + 4);
    int4 aB  = *(const int4*)(ab + bk + 16);
    int4 aB2 = *(const int4*)(ab + bk + 20);
    float4 sA  = *(const float4*)(s2g + j0 + bk);
    float4 sA2 = *(const float4*)(s2g + j0 + bk + 4);
    float4 sB  = *(const float4*)(s2g + j0 + bk + 16);
    float4 sB2 = *(const float4*)(s2g + j0 + bk + 20);

    int   am[16] = { aA.x,aA.y,aA.z,aA.w, aA2.x,aA2.y,aA2.z,aA2.w,
                     aB.x,aB.y,aB.z,aB.w, aB2.x,aB2.y,aB2.z,aB2.w };
    float sm[16] = { sA.x,sA.y,sA.z,sA.w, sA2.x,sA2.y,sA2.z,sA2.w,
                     sB.x,sB.y,sB.z,sB.w, sB2.x,sB2.y,sB2.z,sB2.w };

    float lsum = 0.f;
    v16bf A;
    #pragma unroll
    for (int i = 0; i < 16; ++i) {
      float x = s1v + sm[i];
      float e = fmaxf(x, 0.2f * x);               // leaky_relu(e, 0.2)
      float sc = (am[i] > 0) ? e : NEGV;          // adjacency mask
      float p = __builtin_amdgcn_exp2f(__builtin_fmaf(sc, LOG2E, -MiL));
      lsum += p;
      A[i] = (__bf16)p;                           // A operand, already in layout
    }
    l_run += lsum;

    // P @ WhB : 8 feature tiles, B pre-swizzled (2 x b128 per lane, L2-hot)
    #pragma unroll
    for (int fbv = 0; fbv < 8; ++fbv) {
      v16bf B = *(const v16bf*)(whb + ((size_t)(jb * 8 + fbv) * 32 + lane) * 8);
      acc[fbv] = __builtin_amdgcn_wmma_f32_16x16x32_bf16(false, A, false, B,
                                                         (short)0, acc[fbv],
                                                         false, false);
    }

    if (n + 2 < NJ)
      tdm_load_adj(adj, i0, wave + 8 * (n + 2), &adjbuf[wave][n & 1][0]);
  }

  // combine the two half-row partial sums (kg=0: cols 0-7,16-23 vs kg=1: 8-15,24-31) once
  l_run += __shfl_xor(l_run, 16, 32);

  __syncthreads();   // zeros of Lrow/outacc visible to all before accumulation

  if (kg == 0) atomicAdd(&Lrow[mrow], l_run);
  #pragma unroll
  for (int v = 0; v < 8; ++v) {
    const int row = kg * 8 + v;                   // D rows are kg*8+v
    #pragma unroll
    for (int fbv = 0; fbv < 8; ++fbv)
      atomicAdd(&outacc[row * 128 + fbv * 16 + mrow], acc[fbv][v]);
  }
  __syncthreads();

  // --- normalize + ELU + store ---------------------------------------------
  #pragma unroll
  for (int t = 0; t < 8; ++t) {
    int e = tid + 256 * t;
    int row = e >> 7;
    float v = outacc[e] * __builtin_amdgcn_rcpf(Lrow[row]);
    v = (v > 0.f) ? v : (__builtin_amdgcn_exp2f(v * LOG2E) - 1.f);
    out[(size_t)i0 * 128 + e] = v;
  }
}

// ---------------------------------------------------------------------------
extern "C" void kernel_launch(void* const* d_in, const int* in_sizes, int n_in,
                              void* d_out, int out_size, void* d_ws, size_t ws_size,
                              hipStream_t stream) {
  (void)in_sizes; (void)n_in; (void)out_size; (void)ws_size;
  const float* h   = (const float*)d_in[0];
  const int*   adj = (const int*)d_in[1];
  const float* W   = (const float*)d_in[2];
  const float* a1  = (const float*)d_in[3];
  const float* a2  = (const float*)d_in[4];
  float* out = (float*)d_out;

  float* s1   = (float*)d_ws;                      // 8192 f32
  float* s2   = s1 + 8192;                         // 8192 f32
  float* smax = s2 + 8192;                         // 64 f32 (padding for align)
  unsigned int* whb = (unsigned int*)(smax + 64);  // 256*8*32*8 dwords = 2 MB

  zero_kernel<<<64, 256, 0, stream>>>(s1, 16384);
  wh_kernel<<<512, 256, 0, stream>>>(h, W, a1, a2, s1, s2, whb);
  smax_kernel<<<1, 256, 0, stream>>>(s2, smax);
  gat_kernel<<<512, 256, 0, stream>>>(adj, s1, s2, smax, whb, out);
}
